// Similarities_28132035788951
// MI455X (gfx1250) — matrix operand order repaired and checked by reading, not compiled, not run on hardware
//
#include <hip/hip_runtime.h>
#include <hip/hip_bf16.h>

typedef __attribute__((ext_vector_type(16))) _Float16 v16h;
typedef __attribute__((ext_vector_type(8)))  float    v8f;

namespace {
constexpr int   kH = 384, kW = 512, kB = 128;
constexpr int   kR = 60, kKlen = 121;           // truncate*sigma+0.5 = 60
constexpr float kSigma = 20.0f;
constexpr int   kNFix = 12800;
constexpr size_t kHW  = (size_t)kH * kW;        // 196608
constexpr size_t kBHW = (size_t)kB * kHW;       // 25165824

// ---- workspace layout (bytes) ----
constexpr size_t OFF_FIX  = 0;                         // f32 fix map, reused as emp output
constexpr size_t OFF_TMP  = OFF_FIX + kBHW * 4;        // f16 vertical-blurred intermediate
constexpr size_t OFF_GH   = OFF_TMP + kBHW * 2;        // f16 384x384 band matrix
constexpr size_t OFF_GW   = OFF_GH + (size_t)384 * 384 * 2; // f16 512x512 band matrix
constexpr size_t OFF_GK   = OFF_GW + (size_t)512 * 512 * 2; // f32 gauss taps (121)
constexpr size_t OFF_BSUM = OFF_GK + 512;              // f32 per-batch sums (128)
constexpr size_t OFF_SSUM = OFF_BSUM + 512;            // f32 saliency sum (1)
} // namespace

// ---------------- init: zero fix buffer, accumulators, d_out ----------------
__global__ void k_init(float* __restrict__ fix, float* __restrict__ bsum,
                       float* __restrict__ ssum, float* __restrict__ out) {
  size_t i = (size_t)blockIdx.x * blockDim.x + threadIdx.x;
  size_t stride = (size_t)gridDim.x * blockDim.x;
  for (size_t p = i; p < kBHW; p += stride) fix[p] = 0.0f;
  if (i < (size_t)kB) { bsum[i] = 0.0f; out[i] = 0.0f; }
  if (i == 0) ssum[0] = 0.0f;
}

// ---------------- normalized Gaussian taps ----------------
__global__ void k_gauss(float* __restrict__ gk) {
  __shared__ float w[kKlen];
  __shared__ float s;
  int t = threadIdx.x;
  if (t < kKlen) {
    float x = (float)(t - kR) / kSigma;
    w[t] = __expf(-0.5f * x * x);
  }
  __syncthreads();
  if (t == 0) {
    float acc = 0.0f;
    for (int i = 0; i < kKlen; ++i) acc += w[i];
    s = acc;
  }
  __syncthreads();
  if (t < kKlen) gk[t] = w[t] / s;
}

// ---------------- fill banded Gauss matrices (f16) ----------------
__global__ void k_fill(const float* __restrict__ gk, _Float16* __restrict__ GH,
                       _Float16* __restrict__ GW) {
  int idx = blockIdx.x * blockDim.x + threadIdx.x;
  constexpr int NH = 384 * 384;
  constexpr int NW = 512 * 512;
  if (idx < NH) {
    int i = idx / 384, j = idx % 384, d = i - j;
    GH[idx] = (d >= -kR && d <= kR) ? (_Float16)gk[d + kR] : (_Float16)0.0f;
  } else if (idx < NH + NW) {
    int r = idx - NH;
    int i = r / 512, j = r % 512, d = i - j;
    GW[r] = (d >= -kR && d <= kR) ? (_Float16)gk[d + kR] : (_Float16)0.0f;
  }
}

// ---------------- scatter fixations ----------------
__global__ void k_scatter(const int* __restrict__ ns, const int* __restrict__ ys,
                          const int* __restrict__ xs, float* __restrict__ fix) {
  int i = blockIdx.x * blockDim.x + threadIdx.x;
  if (i < kNFix) {
    size_t p = (size_t)ns[i] * kHW + (size_t)ys[i] * kW + (size_t)xs[i];
    atomicAdd(&fix[p], 1.0f);
  }
}

// ---------------- vertical blur: tmp[n] = G_H x fix[n]  (banded WMMA GEMM) ----
// One wave -> 4 M-tiles (64 rows x 16 cols). The shared operand is the LARGE
// fix buffer (B fragment): loaded once per K step, consumed by 4 WMMAs.
__global__ __launch_bounds__(256) void k_blur_v(const float* __restrict__ fix,
                                                const _Float16* __restrict__ GH,
                                                _Float16* __restrict__ tmp) {
  int wave = blockIdx.x * 8 + (threadIdx.x >> 5);
  int lane = threadIdx.x & 31;
  int tn  = wave & 31;          // 32 tiles along W
  int t2  = wave >> 5;
  int tm4 = t2 % 6;             // 6 groups of 4 tiles along H (24 tiles)
  int n   = t2 / 6;
  if (n >= kB) return;          // wave-uniform; grid sized exactly, never taken
  const int i0 = tm4 * 64, j0 = tn * 16;

  // union K band for rows [i0, i0+63]
  int klo = i0 - kR; if (klo < 0) klo = 0; klo &= ~31;
  int khi = i0 + 63 + kR; if (khi > kH - 1) khi = kH - 1;

  const int mrow = lane & 15;
  const int koff = (lane >> 4) * 8;     // A-fragment K sub-offset per ISA layout
  const float* fixn = fix + (size_t)n * kHW;

  v8f c[4] = {{}, {}, {}, {}};
  for (int kk = klo; kk <= khi; kk += 32) {
    // B: fix[n, kk+K, j0+(lane&15)] -> lane = N, elems 0..7 -> K koff.., 8..15 -> +16 (cvt f32->f16)
    const float* fb = fixn + (size_t)(kk + koff) * kW + j0 + mrow;
    __builtin_prefetch(fb + 32 * kW, 0, 0);
    v16h b;
#pragma unroll
    for (int e = 0; e < 8; ++e) b[e] = (_Float16)fb[(size_t)e * kW];
#pragma unroll
    for (int e = 0; e < 8; ++e) b[8 + e] = (_Float16)fb[(size_t)(16 + e) * kW];
#pragma unroll
    for (int q = 0; q < 4; ++q) {
      // A: G_H[i0+16q+mrow, kk .. kk+31]; elems 0..7 -> K koff.., 8..15 -> +16
      const _Float16* ga = GH + (size_t)(i0 + 16 * q + mrow) * kH + kk;
      v16h a;
#pragma unroll
      for (int e = 0; e < 8; ++e) a[e] = ga[koff + e];
#pragma unroll
      for (int e = 0; e < 8; ++e) a[8 + e] = ga[16 + koff + e];
      c[q] = __builtin_amdgcn_wmma_f32_16x16x32_f16(false, a, false, b,
                                                    (short)0, c[q], false, false);
    }
  }
  // D layout: c[q][v] -> row i0 + 16q + v + (lane>=16 ? 8 : 0), col j0+(lane&15)
  const int msub = (lane >> 4) ? 8 : 0;
  _Float16* tp = tmp + (size_t)n * kHW + (j0 + (lane & 15));
#pragma unroll
  for (int q = 0; q < 4; ++q) {
    const int mbase = i0 + 16 * q + msub;
#pragma unroll
    for (int v = 0; v < 8; ++v) tp[(size_t)(mbase + v) * kW] = (_Float16)c[q][v];
  }
}

// ---------------- horizontal blur: emp[n] = tmp[n] x G_W (banded WMMA GEMM) ----
// One wave -> 4 N-tiles (16 rows x 64 cols). The shared operand is the LARGE
// tmp buffer (A fragment): loaded once per K step, consumed by 4 WMMAs.
// Also accumulates per-batch sums for the normalization.
__global__ __launch_bounds__(256) void k_blur_h(const _Float16* __restrict__ tmp,
                                                const _Float16* __restrict__ GW,
                                                float* __restrict__ emp,
                                                float* __restrict__ bsum) {
  int wave = blockIdx.x * 8 + (threadIdx.x >> 5);
  int lane = threadIdx.x & 31;
  int tn4 = wave & 7;           // 8 groups of 4 tiles along W (32 tiles)
  int t2  = wave >> 3;
  int tm  = t2 % 24;            // 24 tiles along H
  int n   = t2 / 24;
  if (n >= kB) return;
  const int i0 = tm * 16, j0 = tn4 * 64;

  // union K band for cols [j0, j0+63]
  int klo = j0 - kR; if (klo < 0) klo = 0; klo &= ~31;
  int khi = j0 + 63 + kR; if (khi > kW - 1) khi = kW - 1;

  const int mrow = lane & 15;
  const int koff = (lane >> 4) * 8;
  const _Float16* tn_base = tmp + (size_t)n * kHW + (size_t)(i0 + mrow) * kW;

  v8f c[4] = {{}, {}, {}, {}};
  for (int kk = klo; kk <= khi; kk += 32) {
    // A: tmp[n, i0+mrow, kk .. kk+31] -- loaded once, reused by 4 WMMAs
    const _Float16* ta = tn_base + kk;
    __builtin_prefetch(ta + 32, 0, 0);
    v16h a;
#pragma unroll
    for (int e = 0; e < 8; ++e) a[e] = ta[koff + e];
#pragma unroll
    for (int e = 0; e < 8; ++e) a[8 + e] = ta[16 + koff + e];
#pragma unroll
    for (int q = 0; q < 4; ++q) {
      // B: G_W[kk+K, j0+16q+(lane&15)]; elems 0..7 -> K koff.., 8..15 -> +16
      const _Float16* gb = GW + (size_t)(kk + koff) * kW + (j0 + 16 * q + mrow);
      v16h b;
#pragma unroll
      for (int e = 0; e < 8; ++e) b[e] = gb[(size_t)e * kW];
#pragma unroll
      for (int e = 0; e < 8; ++e) b[8 + e] = gb[(size_t)(16 + e) * kW];
      c[q] = __builtin_amdgcn_wmma_f32_16x16x32_f16(false, a, false, b,
                                                    (short)0, c[q], false, false);
    }
  }
  // store f32 emp tiles + reduce wave sum into bsum[n]
  const int msub = (lane >> 4) ? 8 : 0;
  const int mbase = i0 + msub;
  float* en = emp + (size_t)n * kHW;
  float s = 0.0f;
#pragma unroll
  for (int q = 0; q < 4; ++q) {
    float* ep = en + (j0 + 16 * q + (lane & 15));
#pragma unroll
    for (int v = 0; v < 8; ++v) {
      ep[(size_t)(mbase + v) * kW] = c[q][v];
      s += c[q][v];
    }
  }
#pragma unroll
  for (int off = 16; off > 0; off >>= 1) s += __shfl_xor(s, off, 32);
  if (lane == 0) atomicAdd(&bsum[n], s);
}

// ---------------- saliency sum ----------------
__global__ void k_salsum(const float* __restrict__ sal, float* __restrict__ ssum) {
  __shared__ float sd[256];
  int t = threadIdx.x;
  size_t i = (size_t)blockIdx.x * 256 + t;
  sd[t] = (i < kHW) ? sal[i] : 0.0f;
  __syncthreads();
  for (int o = 128; o > 0; o >>= 1) {
    if (t < o) sd[t] += sd[t + o];
    __syncthreads();
  }
  if (t == 0) atomicAdd(ssum, sd[0]);
}

// ---------------- out[n] = sum_p min(emp/bsum, sal/ssum) ----------------
__global__ void k_final(const float* __restrict__ emp, const float* __restrict__ sal,
                        const float* __restrict__ bsum, const float* __restrict__ ssum,
                        float* __restrict__ out) {
  __shared__ float sd[256];
  int t = threadIdx.x;
  int n = blockIdx.x / 96;        // 96 chunks of 2048 pixels = 196608
  int chunk = blockIdx.x % 96;
  float inv_b = 1.0f / bsum[n];
  float inv_s = 1.0f / ssum[0];
  const float* en = emp + (size_t)n * kHW;
  size_t base = (size_t)chunk * 2048;
  float acc = 0.0f;
#pragma unroll
  for (int r = 0; r < 8; ++r) {
    size_t p = base + (size_t)r * 256 + t;
    acc += fminf(en[p] * inv_b, sal[p] * inv_s);
  }
  sd[t] = acc;
  __syncthreads();
  for (int o = 128; o > 0; o >>= 1) {
    if (t < o) sd[t] += sd[t + o];
    __syncthreads();
  }
  if (t == 0) atomicAdd(&out[n], sd[0]);
}

extern "C" void kernel_launch(void* const* d_in, const int* in_sizes, int n_in,
                              void* d_out, int out_size, void* d_ws, size_t ws_size,
                              hipStream_t stream) {
  const float* sal = (const float*)d_in[0];
  const int*   ns  = (const int*)d_in[1];
  const int*   ys  = (const int*)d_in[2];
  const int*   xs  = (const int*)d_in[3];
  float* out = (float*)d_out;

  char* ws = (char*)d_ws;
  float*    fix  = (float*)(ws + OFF_FIX);   // reused as emp after pass V consumes it
  _Float16* tmp  = (_Float16*)(ws + OFF_TMP);
  _Float16* GH   = (_Float16*)(ws + OFF_GH);
  _Float16* GW   = (_Float16*)(ws + OFF_GW);
  float*    gk   = (float*)(ws + OFF_GK);
  float*    bsum = (float*)(ws + OFF_BSUM);
  float*    ssum = (float*)(ws + OFF_SSUM);

  // 1. zero fix map, accumulators, output
  k_init<<<4096, 256, 0, stream>>>(fix, bsum, ssum, out);
  // 2. Gaussian taps
  k_gauss<<<1, 128, 0, stream>>>(gk);
  // 3. banded Gauss matrices (384*384 + 512*512 = 409600 entries)
  k_fill<<<1600, 256, 0, stream>>>(gk, GH, GW);
  // 4. scatter fixations
  k_scatter<<<(kNFix + 255) / 256, 256, 0, stream>>>(ns, ys, xs, fix);
  // 5. vertical banded GEMM: 128*6*32 = 24576 waves (4 M-tiles each)
  k_blur_v<<<3072, 256, 0, stream>>>(fix, GH, tmp);
  // 6. horizontal banded GEMM: 128*24*8 = 24576 waves (4 N-tiles each)
  k_blur_h<<<3072, 256, 0, stream>>>(tmp, GW, fix, bsum);
  // 7. saliency normalization sum (196608 / 256 = 768 blocks)
  k_salsum<<<768, 256, 0, stream>>>(sal, ssum);
  // 8. per-batch min-sum
  k_final<<<kB * 96, 256, 0, stream>>>(fix, sal, bsum, ssum, out);

  (void)in_sizes; (void)n_in; (void)out_size; (void)ws_size;
}
